// RPN_75926431858962
// MI455X (gfx1250) — compile-verified, matrix operand order checked by reading, not run on hardware
//
#include <hip/hip_runtime.h>
#include <hip/hip_bf16.h>
#include <math.h>
#include <stdint.h>

// ---------------------------------------------------------------------------
// CDNA5 (gfx1250) RPN:
//   * conv3x3(256->256) as implicit GEMM on v_wmma_f32_16x16x32_bf16
//   * A/B tiles staged into LDS by the Tensor Data Mover (tensor_load_to_lds,
//     double buffered; issue-next-then-wait(<=2) exploits in-order TENSORcnt)
//   * fused bias+ReLU+1x1 heads in LDS, then topk-select + decode + NMS
// ---------------------------------------------------------------------------

typedef __attribute__((ext_vector_type(16))) __bf16 v16bf;
typedef __attribute__((ext_vector_type(8)))  float  v8f;
typedef __attribute__((ext_vector_type(4)))  unsigned int u32x4;
typedef __attribute__((ext_vector_type(8)))  int    i32x8;
typedef __attribute__((ext_vector_type(4)))  int    i32x4;

struct Frag32 { uint4 lo; uint4 hi; };   // 32 bytes == one v16bf operand

__device__ __forceinline__ unsigned short f2bf(float f) {
    unsigned int u = __builtin_bit_cast(unsigned int, f);
    unsigned int r = u + 0x7FFFu + ((u >> 16) & 1u);   // round-to-nearest-even
    return (unsigned short)(r >> 16);
}
__device__ __forceinline__ float bf2f(unsigned short s) {
    unsigned int u = ((unsigned int)s) << 16;
    return __builtin_bit_cast(float, u);
}
__device__ __forceinline__ unsigned fkey(float f) {
    unsigned u = __builtin_bit_cast(unsigned, f);
    return (u & 0x80000000u) ? ~u : (u | 0x80000000u);  // order-preserving key
}

#define ATOT      261888
#define CAP       6000
#define MCAND     30000
#define POSTK     1000
#define NEG_INF   (-__builtin_inff())
#define SCALE_CLAMP 4.1351666f   // log(1000/16)
#define KTOT      2304           // 9 * 256
#define GUARD_EL  131072         // halo guard elements (bf16) each side

// LDS layout (bytes) inside one 56KB block
#define AS_OFF(b) ((b) ? 4096u : 0u)        // 64px x 32k bf16  = 4KB each
#define BS_OFF(b) ((b) ? 24576u : 8192u)    // 256oc x 32k bf16 = 16KB each
#define TS_OFF    0u                         // 64px x 256oc bf16 = 32KB (epilogue)
#define WH_OFF    40960u                     // 15 x 256 f32     = 15KB
#define SMEM_BYTES 57344

__device__ __forceinline__ int atot_ofs(int lvl) {
    int o = 0;
    for (int l = 0; l < lvl; ++l) { int w = 256 >> l; o += 3 * w * w; }
    return o;
}

// ---- Tensor Data Mover: 2-D bf16 tile (tile0 x tile1 rows) -> LDS ----------
__device__ __forceinline__ void tdm_load_2d(unsigned lds_off, const void* gptr,
                                            unsigned tile0, unsigned tile1,
                                            unsigned td0, unsigned td1,
                                            unsigned long long stride0)
{
    unsigned long long ga = (unsigned long long)(uintptr_t)gptr;
    u32x4 g0;
    g0[0] = 1u;                                               // count=1, load
    g0[1] = lds_off;                                          // LDS byte addr
    g0[2] = (unsigned)(ga & 0xFFFFFFFFu);                     // global[31:0]
    g0[3] = (unsigned)((ga >> 32) & 0x01FFFFFFu) | (2u << 30);// global[56:32]|type=2
    i32x8 g1;
    g1[0] = (int)(1u << 16);                                  // data_size=2B
    g1[1] = (int)((td0 & 0xFFFFu) << 16);                     // tensor_dim0[15:0]
    g1[2] = (int)((td0 >> 16) | ((td1 & 0xFFFFu) << 16));     // td0[31:16]|td1[15:0]
    g1[3] = (int)((td1 >> 16) | (tile0 << 16));               // td1[31:16]|tile_dim0
    g1[4] = (int)(tile1 & 0xFFFFu);                           // tile_dim1 (tile_dim2=0)
    g1[5] = (int)(stride0 & 0xFFFFFFFFull);                   // dim0_stride[31:0]
    g1[6] = (int)((stride0 >> 32) & 0xFFFFull);               // dim0_stride[47:32]
    g1[7] = 0;
    i32x4 z4; z4[0] = z4[1] = z4[2] = z4[3] = 0;              // groups 2/3 unused (2-D)
    i32x8 z8; for (int e = 0; e < 8; ++e) z8[e] = 0;
    __builtin_amdgcn_tensor_load_to_lds(g0, g1, z4, z4, z8, 0);
}

// ---------------------------------------------------------------------------
// Repack kernels (run once per launch; bandwidth-trivial at 23.3 TB/s)
// ---------------------------------------------------------------------------
__global__ __launch_bounds__(256)
void repack_weights(const float* __restrict__ conv_w, unsigned short* __restrict__ wbf)
{
    int g = blockIdx.x * 256 + threadIdx.x;          // oc*2304 + k
    if (g >= 256 * KTOT) return;
    int oc = g / KTOT, k = g - oc * KTOT;
    int c = k & 255, kidx = k >> 8;
    wbf[g] = f2bf(conv_w[(oc * 256 + c) * 9 + kidx]);
}

__global__ __launch_bounds__(256)
void repack_feats(const float* __restrict__ fin, unsigned short* __restrict__ fb,
                  int HW)
{
    long g = (long)blockIdx.x * 256 + threadIdx.x;   // ((n*HW)+hw)*256 + c
    if (g >= 2L * HW * 256) return;
    int  c   = (int)(g & 255);
    long phw = g >> 8;
    int  hw  = (int)(phw % HW);
    int  n   = (int)(phw / HW);
    fb[g] = f2bf(fin[((long)(n * 256 + c)) * HW + hw]);
}

__global__ __launch_bounds__(256)
void fill_zero_u16(unsigned short* __restrict__ p, long n)
{
    long g = (long)blockIdx.x * 256 + threadIdx.x;
    if (g < n) p[g] = 0;
}

// ---------------------------------------------------------------------------
// Kernel A: conv3x3 + bias + ReLU + fused 1x1 heads.
// Block: 256 threads (8 wave32). Tile: 64 pixels x 256 out-channels.
// FEAT_TDM=true : A tile via TDM from repacked bf16 NHWC features.
// FEAT_TDM=false: A tile staged by VALU from fp32 NCHW (workspace fallback);
//                 B tile always via TDM from repacked bf16 weights.
// ---------------------------------------------------------------------------
template <bool FEAT_TDM>
__global__ __launch_bounds__(256)
void rpn_conv_fused(const unsigned short* __restrict__ fbf,   // bf16 NHWC (or null)
                    const float* __restrict__ fin,            // fp32 NCHW
                    const unsigned short* __restrict__ wbf,   // bf16 [256][2304]
                    const float* __restrict__ conv_b,
                    const float* __restrict__ obj_w,
                    const float* __restrict__ obj_b,
                    const float* __restrict__ delta_w,
                    const float* __restrict__ delta_b,
                    float* __restrict__ combined,
                    int H, int W, int aofs)
{
    __shared__ __align__(16) unsigned char smem[SMEM_BYTES];

    const int tid   = threadIdx.x;
    const int lane  = tid & 31;
    const int wave  = tid >> 5;
    const int waveM = wave >> 2;        // 0..1 : 32 pixels
    const int waveN = wave & 3;         // 0..3 : 64 out-channels
    const int half  = lane >> 4;
    const int l16   = lane & 15;
    const int j0    = half * 8;

    const int n      = blockIdx.y;
    const int pxBase = blockIdx.x * 64;
    const int HW     = H * W;

    const unsigned short* fb_n = FEAT_TDM ? (fbf + (long)n * HW * 256) : nullptr;
    const unsigned asBase = (unsigned)(size_t)(smem);        // LDS byte offset of smem

    // fallback path: per-thread pixel coords hoisted out of the K loop
    int hs[8], wss[8];
    if (!FEAT_TDM) {
        #pragma unroll
        for (int j = 0; j < 8; ++j) {
            int px = pxBase + (tid >> 5) + 8 * j;
            hs[j] = px / W; wss[j] = px - hs[j] * W;
        }
    }

    v8f z; for (int e = 0; e < 8; ++e) z[e] = 0.0f;
    v8f acc[2][4];
    for (int mt = 0; mt < 2; ++mt)
        for (int nt = 0; nt < 4; ++nt) acc[mt][nt] = z;

    // ---- preload chunk 0 (wave0 only; TENSORcnt tracks completion) --------
    if (tid < 32) {
        if (FEAT_TDM) {
            int shift = -W - 1;                       // ky=0,kx=0 tap
            tdm_load_2d(asBase + AS_OFF(0), fb_n + ((long)(pxBase + shift) * 256),
                        32, 64, 256, 0x40000000u, 256);
        }
        tdm_load_2d(asBase + BS_OFF(0), wbf, 32, 256, KTOT, 256, KTOT);
    }

    for (int i = 0; i < KTOT / 32; ++i) {
        const int k0  = i * 32;
        const int cur = i & 1;
        const bool more = (i + 1) < (KTOT / 32);

        __syncthreads();     // everyone done computing from buffer 1-cur

        if (!FEAT_TDM) {
            // stage A tile (64px x 32c) from fp32 NCHW; fixed tap per chunk
            int c0 = k0 & 255, kidx = k0 >> 8;
            int ky = kidx / 3, kx = kidx - 3 * ky;
            int c = c0 + (tid & 31);
            const float* fc = fin + ((long)(n * 256 + c)) * HW;
            unsigned short* As = (unsigned short*)(smem + AS_OFF(0));
            #pragma unroll
            for (int j = 0; j < 8; ++j) {
                int hh = hs[j] + ky - 1, ww = wss[j] + kx - 1;
                float v = (hh >= 0 && hh < H && ww >= 0 && ww < W)
                              ? fc[hh * W + ww] : 0.0f;
                As[((tid >> 5) + 8 * j) * 32 + (tid & 31)] = f2bf(v);
            }
        }
        if (tid < 32) {
            if (more) {
                // issue BOTH next-chunk loads first, then wait: with in-order
                // TENSORcnt retirement, wait<=outstanding_next proves the
                // current chunk's tiles have fully landed in LDS.
                int kn = k0 + 32;
                if (FEAT_TDM) {
                    int c0 = kn & 255, kidx = kn >> 8;
                    int ky = kidx / 3, kx = kidx - 3 * ky;
                    int shift = (ky - 1) * W + (kx - 1);
                    tdm_load_2d(asBase + AS_OFF(1 - cur),
                                fb_n + ((long)(pxBase + shift) * 256 + c0),
                                32, 64, 256, 0x40000000u, 256);
                }
                tdm_load_2d(asBase + BS_OFF(1 - cur), wbf + kn,
                            32, 256, KTOT, 256, KTOT);
                if (FEAT_TDM) __builtin_amdgcn_s_wait_tensorcnt(2);
                else          __builtin_amdgcn_s_wait_tensorcnt(1);
            } else {
                __builtin_amdgcn_s_wait_tensorcnt(0);
            }
        }
        __syncthreads();     // staged data visible to all waves

        const unsigned short* As =
            (const unsigned short*)(smem + (FEAT_TDM ? AS_OFF(cur) : AS_OFF(0)));
        const unsigned short* Bs = (const unsigned short*)(smem + BS_OFF(cur));

        v16bf afr[2];
        #pragma unroll
        for (int mt = 0; mt < 2; ++mt) {
            int row = waveM * 32 + mt * 16 + l16;
            Frag32 f;
            f.lo = *(const uint4*)(As + row * 32 + j0);
            f.hi = *(const uint4*)(As + row * 32 + 16 + j0);
            afr[mt] = __builtin_bit_cast(v16bf, f);
        }
        #pragma unroll
        for (int nt = 0; nt < 4; ++nt) {
            int oc = waveN * 64 + nt * 16 + l16;
            Frag32 f;
            f.lo = *(const uint4*)(Bs + oc * 32 + j0);
            f.hi = *(const uint4*)(Bs + oc * 32 + 16 + j0);
            v16bf bfr = __builtin_bit_cast(v16bf, f);
            #pragma unroll
            for (int mt = 0; mt < 2; ++mt)
                acc[mt][nt] = __builtin_amdgcn_wmma_f32_16x16x32_bf16(
                    false, afr[mt], false, bfr, (short)0, acc[mt][nt],
                    false, false);
        }
    }

    __syncthreads();   // conv buffers dead; reuse region as Ts

    unsigned short* Ts = (unsigned short*)(smem + TS_OFF);   // [64][256] bf16
    float*          Wh = (float*)(smem + WH_OFF);            // [15][256]

    for (int e = tid; e < 15 * 256; e += 256) {
        int j = e >> 8, c = e & 255;
        Wh[e] = (j < 3) ? obj_w[j * 256 + c] : delta_w[(j - 3) * 256 + c];
    }
    #pragma unroll
    for (int mt = 0; mt < 2; ++mt)
        #pragma unroll
        for (int nt = 0; nt < 4; ++nt) {
            int oc = waveN * 64 + nt * 16 + l16;
            float bias = conv_b[oc];
            #pragma unroll
            for (int v = 0; v < 8; ++v) {
                int p = waveM * 32 + mt * 16 + v + 8 * half;
                float t = acc[mt][nt][v] + bias;
                t = t > 0.0f ? t : 0.0f;
                Ts[p * 256 + oc] = f2bf(t);
            }
        }
    __syncthreads();

    if (tid < 64) {
        int px = pxBase + tid;
        if (px < HW) {
            float o[15];
            #pragma unroll
            for (int j = 0; j < 15; ++j) o[j] = (j < 3) ? obj_b[j] : delta_b[j - 3];
            for (int c = 0; c < 256; ++c) {
                float tv = bf2f(Ts[tid * 256 + c]);
                #pragma unroll
                for (int j = 0; j < 15; ++j) o[j] += tv * Wh[j * 256 + c];
            }
            long base = ((long)n * ATOT + aofs + (long)px * 3) * 5;
            #pragma unroll
            for (int a = 0; a < 3; ++a) {
                combined[base + a * 5 + 0] = o[a];
                #pragma unroll
                for (int i2 = 0; i2 < 4; ++i2)
                    combined[base + a * 5 + 1 + i2] = o[3 + a * 4 + i2];
            }
        }
    }
}

// ---------------------------------------------------------------------------
// Kernel B: per-(image,level) top-6000 select via 2048-bin radix histogram.
// ---------------------------------------------------------------------------
__global__ __launch_bounds__(256)
void rpn_select(const float* __restrict__ combined,
                int* __restrict__ cand_idx, float* __restrict__ cand_score)
{
    __shared__ int hist[2048];
    __shared__ int sTB, sNA, sRem, ctrA, ctrE;

    const int lvl = blockIdx.x, n = blockIdx.y, tid = threadIdx.x;
    const int Wl  = 256 >> lvl;
    const int HWA = 3 * Wl * Wl;
    const int ofs = atot_ofs(lvl);
    const float* lg = combined + ((long)n * ATOT + ofs) * 5;
    int*   ci = cand_idx   + (n * 5 + lvl) * CAP;
    float* cs = cand_score + (n * 5 + lvl) * CAP;

    for (int s = tid; s < CAP; s += 256) { ci[s] = -1; cs[s] = NEG_INF; }

    if (HWA <= CAP) {
        for (int s = tid; s < HWA; s += 256) { ci[s] = s; cs[s] = lg[(long)s * 5]; }
        return;
    }
    for (int b = tid; b < 2048; b += 256) hist[b] = 0;
    __syncthreads();
    for (int s = tid; s < HWA; s += 256)
        atomicAdd(&hist[fkey(lg[(long)s * 5]) >> 21], 1);
    __syncthreads();
    if (tid == 0) {
        int cum = 0, b;
        for (b = 2047; b >= 0; --b) { if (cum + hist[b] >= CAP) break; cum += hist[b]; }
        sTB = b; sNA = cum; sRem = CAP - cum; ctrA = 0; ctrE = 0;
    }
    __syncthreads();
    const int tb = sTB, nAb = sNA, rem = sRem;
    for (int s = tid; s < HWA; s += 256) {
        float v = lg[(long)s * 5];
        int b = (int)(fkey(v) >> 21);
        if (b > tb) {
            int pos = atomicAdd(&ctrA, 1);
            ci[pos] = s; cs[pos] = v;
        } else if (b == tb) {
            int pos = atomicAdd(&ctrE, 1);
            if (pos < rem) { ci[nAb + pos] = s; cs[nAb + pos] = v; }
        }
    }
}

// ---------------------------------------------------------------------------
// Kernel C: decode candidates, clip, level offset for batched NMS.
// ---------------------------------------------------------------------------
__global__ __launch_bounds__(256)
void rpn_decode(const float* __restrict__ combined,
                const int* __restrict__ cand_idx,
                const float* __restrict__ cand_score,
                float* __restrict__ s_mut,
                float* __restrict__ cbox, float* __restrict__ obox)
{
    int g = blockIdx.x * 256 + threadIdx.x;
    if (g >= 2 * MCAND) return;
    int n   = g / MCAND;
    int r   = g - n * MCAND;
    int lvl = r / CAP;

    int   idx   = cand_idx[g];
    float score = cand_score[g];
    s_mut[g] = score;

    float b0 = 0.f, b1 = 0.f, b2 = 0.f, b3 = 0.f;
    if (idx >= 0) {
        int Wl = 256 >> lvl, stride = 4 << lvl, size = 32 << lvl;
        int a  = idx % 3, hw = idx / 3;
        int h  = hw / Wl, w = hw - h * Wl;
        float rr = (a == 0) ? 0.5f : (a == 1 ? 1.0f : 2.0f);
        float sr = sqrtf(rr);
        float wa = (float)size / sr;
        float ha = (float)size * sr;
        float cx = (float)(w * stride);
        float cy = (float)(h * stride);

        long  db = ((long)n * ATOT + atot_ofs(lvl) + idx) * 5;
        float dx = combined[db + 1], dy = combined[db + 2];
        float dw = fminf(combined[db + 3], SCALE_CLAMP);
        float dh = fminf(combined[db + 4], SCALE_CLAMP);
        float pcx = dx * wa + cx, pcy = dy * ha + cy;
        float pw  = expf(dw) * wa, ph = expf(dh) * ha;

        b0 = fminf(fmaxf(pcx - 0.5f * pw, 0.f), 1024.f);
        b1 = fminf(fmaxf(pcy - 0.5f * ph, 0.f), 1024.f);
        b2 = fminf(fmaxf(pcx + 0.5f * pw, 0.f), 1024.f);
        b3 = fminf(fmaxf(pcy + 0.5f * ph, 0.f), 1024.f);
    }
    float off = (float)lvl * 2000.0f;
    cbox[g * 4 + 0] = b0; cbox[g * 4 + 1] = b1;
    cbox[g * 4 + 2] = b2; cbox[g * 4 + 3] = b3;
    obox[g * 4 + 0] = b0 + off; obox[g * 4 + 1] = b1 + off;
    obox[g * 4 + 2] = b2 + off; obox[g * 4 + 3] = b3 + off;
}

// ---------------------------------------------------------------------------
// Kernel D: greedy NMS, 1000 fixed iterations, one block per image.
// ---------------------------------------------------------------------------
__global__ __launch_bounds__(256)
void rpn_nms(const float* __restrict__ s0, float* __restrict__ s,
             const float* __restrict__ cbox, const float* __restrict__ obox,
             float* __restrict__ keep_boxes, float* __restrict__ keep_scores,
             float* __restrict__ valid)
{
    __shared__ float rv[256];
    __shared__ int   ri[256];
    __shared__ float bx[4];

    const int n = blockIdx.x, tid = threadIdx.x;
    const float* S0 = s0   + (long)n * MCAND;
    float*       Sm = s    + (long)n * MCAND;
    const float* CB = cbox + (long)n * MCAND * 4;
    const float* OB = obox + (long)n * MCAND * 4;

    for (int t = 0; t < POSTK; ++t) {
        float bv = NEG_INF; int bidx = 0;
        for (int j = tid; j < MCAND; j += 256) {
            float v = Sm[j];
            if (v > bv || (v == bv && j < bidx)) { bv = v; bidx = j; }
        }
        rv[tid] = bv; ri[tid] = bidx;
        __syncthreads();
        for (int off = 128; off > 0; off >>= 1) {
            if (tid < off) {
                float v2 = rv[tid + off]; int i2 = ri[tid + off];
                if (v2 > rv[tid] || (v2 == rv[tid] && i2 < ri[tid])) {
                    rv[tid] = v2; ri[tid] = i2;
                }
            }
            __syncthreads();
        }
        if (tid == 0) {
            int i = ri[0];
            keep_scores[n * POSTK + t] = S0[i];
            valid[n * POSTK + t] = (rv[0] > NEG_INF) ? 1.0f : 0.0f;
            for (int q = 0; q < 4; ++q) {
                keep_boxes[(n * POSTK + t) * 4 + q] = CB[i * 4 + q];
                bx[q] = OB[i * 4 + q];
            }
            Sm[i] = NEG_INF;
        }
        __syncthreads();
        const float ax1 = bx[0], ay1 = bx[1], ax2 = bx[2], ay2 = bx[3];
        const float aarea = (ax2 - ax1) * (ay2 - ay1);
        for (int j = tid; j < MCAND; j += 256) {
            if (Sm[j] == NEG_INF) continue;
            float jx1 = OB[j * 4], jy1 = OB[j * 4 + 1];
            float jx2 = OB[j * 4 + 2], jy2 = OB[j * 4 + 3];
            float iw = fmaxf(fminf(ax2, jx2) - fmaxf(ax1, jx1), 0.0f);
            float ih = fmaxf(fminf(ay2, jy2) - fmaxf(ay1, jy1), 0.0f);
            float inter = iw * ih;
            float areaj = (jx2 - jx1) * (jy2 - jy1);
            float iou = inter / (aarea + areaj - inter + 1e-9f);
            if (iou > 0.7f) Sm[j] = NEG_INF;
        }
        __syncthreads();
    }
}

// ---------------------------------------------------------------------------
extern "C" void kernel_launch(void* const* d_in, const int* in_sizes, int n_in,
                              void* d_out, int out_size, void* d_ws, size_t ws_size,
                              hipStream_t stream)
{
    const float* feats[5] = {
        (const float*)d_in[0], (const float*)d_in[1], (const float*)d_in[2],
        (const float*)d_in[3], (const float*)d_in[4] };
    const float* conv_w  = (const float*)d_in[5];
    const float* conv_b  = (const float*)d_in[6];
    const float* obj_w   = (const float*)d_in[7];
    const float* obj_b   = (const float*)d_in[8];
    const float* delta_w = (const float*)d_in[9];
    const float* delta_b = (const float*)d_in[10];

    float* out        = (float*)d_out;
    float* combined   = out;                                  // [2, ATOT, 5]
    float* keep_boxes = out + 2L * ATOT * 5;                  // [2, 1000, 4]
    float* keep_score = keep_boxes + 2L * POSTK * 4;          // [2, 1000]
    float* validf     = keep_score + 2L * POSTK;              // [2, 1000]

    char* ws = (char*)d_ws;
    size_t o = 0;
    auto alloc = [&](size_t bytes) {
        void* p = ws + o;
        o = (o + bytes + 255) & ~(size_t)255;
        return p;
    };
    int*   cand_idx   = (int*)  alloc(2L * MCAND * sizeof(int));
    float* cand_score = (float*)alloc(2L * MCAND * sizeof(float));
    float* s_mut      = (float*)alloc(2L * MCAND * sizeof(float));
    float* cbox       = (float*)alloc(2L * MCAND * 4 * sizeof(float));
    float* obox       = (float*)alloc(2L * MCAND * 4 * sizeof(float));
    unsigned short* wbf = (unsigned short*)alloc(256L * KTOT * 2);

    static const int Hs[5]  = {256, 128, 64, 32, 16};
    static const int Ofs[5] = {0, 196608, 245760, 258048, 261120};

    // Can we afford bf16-NHWC repacked features (guarded for the 3x3 halo)?
    size_t featNeed = 0;
    for (int l = 0; l < 5; ++l) {
        long HW = (long)Hs[l] * Hs[l];
        featNeed += ((2L * HW * 256 + 2L * GUARD_EL) * 2 + 256);
    }
    const bool featTDM = (ws_size >= o + featNeed);

    unsigned short* fb[5] = {nullptr, nullptr, nullptr, nullptr, nullptr};
    if (featTDM) {
        for (int l = 0; l < 5; ++l) {
            long HW = (long)Hs[l] * Hs[l];
            unsigned short* raw =
                (unsigned short*)alloc((2L * HW * 256 + 2L * GUARD_EL) * 2);
            fb[l] = raw + GUARD_EL;   // data start; guards on both sides
        }
    }

    // ---- one-time repacks ------------------------------------------------
    repack_weights<<<dim3((256 * KTOT + 255) / 256), 256, 0, stream>>>(conv_w, wbf);
    if (featTDM) {
        for (int l = 0; l < 5; ++l) {
            long HW = (long)Hs[l] * Hs[l];
            long nel = 2L * HW * 256;
            fill_zero_u16<<<dim3((GUARD_EL + 255) / 256), 256, 0, stream>>>(
                fb[l] - GUARD_EL, GUARD_EL);
            fill_zero_u16<<<dim3((GUARD_EL + 255) / 256), 256, 0, stream>>>(
                fb[l] + nel, GUARD_EL);
            repack_feats<<<dim3((unsigned)((nel + 255) / 256)), 256, 0, stream>>>(
                feats[l], fb[l], (int)HW);
        }
    }

    // ---- fused conv + heads ---------------------------------------------
    for (int l = 0; l < 5; ++l) {
        int H = Hs[l], W = Hs[l];
        int tiles = (H * W) / 64;
        if (featTDM)
            rpn_conv_fused<true><<<dim3(tiles, 2), 256, 0, stream>>>(
                fb[l], feats[l], wbf, conv_b, obj_w, obj_b, delta_w, delta_b,
                combined, H, W, Ofs[l]);
        else
            rpn_conv_fused<false><<<dim3(tiles, 2), 256, 0, stream>>>(
                nullptr, feats[l], wbf, conv_b, obj_w, obj_b, delta_w, delta_b,
                combined, H, W, Ofs[l]);
    }

    rpn_select<<<dim3(5, 2), 256, 0, stream>>>(combined, cand_idx, cand_score);
    rpn_decode<<<dim3((2 * MCAND + 255) / 256), 256, 0, stream>>>(
        combined, cand_idx, cand_score, s_mut, cbox, obox);
    rpn_nms<<<dim3(2), 256, 0, stream>>>(
        cand_score, s_mut, cbox, obox, keep_boxes, keep_score, validf);
}
